// GNN_28741921145294
// MI455X (gfx1250) — matrix operand order repaired and checked by reading, not compiled, not run on hardware
//
#include <hip/hip_runtime.h>
#include <hip/hip_bf16.h>

typedef __attribute__((ext_vector_type(16))) _Float16 v16h;
typedef __attribute__((ext_vector_type(8)))  float    v8f;

#define FD 64   // feature dim

// ---------------- zero-init scratch (runs every call; graph-replay safe) ---
__global__ void gnn_zero_kernel(float* __restrict__ p, long long count) {
  long long i = (long long)blockIdx.x * blockDim.x + threadIdx.x;
  long long stride = (long long)gridDim.x * blockDim.x;
  for (; i < count; i += stride) p[i] = 0.0f;
}

// ---------------- fused 4-way projection: o_k = x @ W_k + b_k  (WMMA) -------
// Block = 128 threads = 4 wave32s; wave w computes projection w for one
// 16-row node tile. Per wave: 4 N-tiles x 2 K-chunks = 8 v_wmma_f32_16x16x32_f16.
__global__ void __launch_bounds__(128)
gnn_proj_kernel(const float* __restrict__ x,
                const float* __restrict__ w1, const float* __restrict__ b1,
                const float* __restrict__ w2, const float* __restrict__ b2,
                const float* __restrict__ w3, const float* __restrict__ b3,
                const float* __restrict__ w4, const float* __restrict__ b4,
                float* __restrict__ o1, float* __restrict__ o2,
                float* __restrict__ o3, float* __restrict__ o4,
                int n) {
  const int wave = threadIdx.x >> 5;
  const int lane = threadIdx.x & 31;
  const int row0 = blockIdx.x * 16;
  const bool fullTile = (row0 + 16) <= n;   // uniform: all tiles but the last

  const float* W; const float* Bv; float* O;
  if      (wave == 0) { W = w1; Bv = b1; O = o1; }
  else if (wave == 1) { W = w2; Bv = b2; O = o2; }
  else if (wave == 2) { W = w3; Bv = b3; O = o3; }
  else                { W = w4; Bv = b4; O = o4; }

  // ---- A fragment (16x32 f16, per ISA layout) for K-chunks 0 and 1 ----
  // lane L<16: row M=L, halves 0..7 = K 0..7, halves 8..15 = K 16..23
  // lane L>=16: row M=L-16, halves 0..7 = K 8..15, halves 8..15 = K 24..31
  const int m      = lane & 15;
  const int khalf  = (lane >> 4) * 8;
  const int row    = row0 + m;
  const float* arow = x + (size_t)row * FD;

  v16h a0, a1;
  if (fullTile) {
    #pragma unroll
    for (int i = 0; i < 16; ++i) {
      int k = khalf + i + ((i >= 8) ? 8 : 0);
      a0[i] = (_Float16)arow[k];
      a1[i] = (_Float16)arow[k + 32];
    }
  } else {
    const bool rowok = row < n;
    #pragma unroll
    for (int i = 0; i < 16; ++i) {
      int k = khalf + i + ((i >= 8) ? 8 : 0);
      a0[i] = rowok ? (_Float16)arow[k]      : (_Float16)0.0f;
      a1[i] = rowok ? (_Float16)arow[k + 32] : (_Float16)0.0f;
    }
  }

  // ---- B fragment (32x16 f16): lane covers col n=lane&15, halves cover K ----
  const int ncolLane = lane & 15;
  const int kb       = (lane >> 4) * 8;
  const int mbase    = (lane >> 4) * 8;

  #pragma unroll
  for (int nt = 0; nt < 4; ++nt) {
    const int ncol = nt * 16 + ncolLane;
    v16h bf0, bf1;
    #pragma unroll
    for (int i = 0; i < 16; ++i) {
      bf0[i] = (_Float16)W[(kb + i + ((i >= 8) ? 8 : 0)) * FD + ncol];
      bf1[i] = (_Float16)W[(32 + kb + i + ((i >= 8) ? 8 : 0)) * FD + ncol];
    }
    v8f c;
    const float bias = Bv[ncol];
    #pragma unroll
    for (int i = 0; i < 8; ++i) c[i] = bias;
    c = __builtin_amdgcn_wmma_f32_16x16x32_f16(false, a0, false, bf0, (short)0, c, false, false);
    c = __builtin_amdgcn_wmma_f32_16x16x32_f16(false, a1, false, bf1, (short)0, c, false, false);
    // C/D layout: lane n = lane&15; float r -> row m = r + (lane>>4)*8
    float* obase = O + (size_t)(row0 + mbase) * FD + ncol;
    if (fullTile) {
      #pragma unroll
      for (int r = 0; r < 8; ++r) obase[(size_t)r * FD] = c[r];
    } else {
      #pragma unroll
      for (int r = 0; r < 8; ++r)
        if (row0 + mbase + r < n) obase[(size_t)r * FD] = c[r];
    }
  }
}

// ---------------- per-node scalars: q5 = x3.w5_w, qb = x3.w5_b --------------
__global__ void gnn_q_kernel(const float* __restrict__ x3,
                             const float* __restrict__ w5,
                             const float* __restrict__ b5,
                             float* __restrict__ q5, float* __restrict__ qb,
                             int n) {
  int i = blockIdx.x * blockDim.x + threadIdx.x;
  if (i >= n) return;
  const float* r = x3 + (size_t)i * FD;
  float s5 = 0.0f, sb = 0.0f;
  #pragma unroll
  for (int d = 0; d < FD; ++d) { float v = r[d]; s5 += v * w5[d]; sb += v * b5[d]; }
  q5[i] = s5; qb[i] = sb;
}

__device__ __forceinline__ unsigned f32_to_ordkey(float f) {
  unsigned b = __float_as_uint(f);
  return (b & 0x80000000u) ? ~b : (b | 0x80000000u);
}
__device__ __forceinline__ float ordkey_to_f32(unsigned k) {
  unsigned b = (k & 0x80000000u) ? (k ^ 0x80000000u) : ~k;
  return __uint_as_float(b);
}

// ---------------- pass A: per-edge score + segment max (wave per edge) ------
__global__ void gnn_score_kernel(const float* __restrict__ x3,
                                 const float* __restrict__ x4,
                                 const float* __restrict__ ea,
                                 const float* __restrict__ q5,
                                 const float* __restrict__ qb,
                                 const int* __restrict__ src,
                                 const int* __restrict__ dst,
                                 float* __restrict__ scores,
                                 unsigned* __restrict__ mkey,
                                 int E) {
  int wid  = (int)((blockIdx.x * (long long)blockDim.x + threadIdx.x) >> 5);
  int lane = threadIdx.x & 31;
  if (wid >= E) return;
  int s = src[wid], d = dst[wid];
  const float2* a = (const float2*)(x3 + (size_t)d * FD);
  const float2* b = (const float2*)(x4 + (size_t)s * FD);
  float2 av = a[lane], bv = b[lane];
  float p = av.x * bv.x + av.y * bv.y;
  #pragma unroll
  for (int off = 16; off > 0; off >>= 1) p += __shfl_xor(p, off, 32);
  float sc = (p + ea[wid] * q5[d] + qb[d]) * 0.125f;   // / sqrt(64)
  if (lane == 0) {
    scores[wid] = sc;
    atomicMax(&mkey[d], f32_to_ordkey(sc));
  }
}

// ---------------- pass B: exp(score - max) + segment sum --------------------
__global__ void gnn_exp_kernel(const int* __restrict__ dst,
                               const unsigned* __restrict__ mkey,
                               float* __restrict__ scores,   // in: score, out: exp
                               float* __restrict__ den, int E) {
  int e = blockIdx.x * blockDim.x + threadIdx.x;
  if (e >= E) return;
  int d = dst[e];
  float mx = ordkey_to_f32(mkey[d]);
  float ex = __expf(scores[e] - mx);
  scores[e] = ex;
  atomicAdd(&den[d], ex);
}

// ---------------- pass C: scatter alpha * x2[src] (wave per edge) -----------
__global__ void gnn_scatter_kernel(const float* __restrict__ x2,
                                   const float* __restrict__ ex,
                                   const float* __restrict__ den,
                                   const int* __restrict__ src,
                                   const int* __restrict__ dst,
                                   float* __restrict__ aggr, int E) {
  int wid  = (int)((blockIdx.x * (long long)blockDim.x + threadIdx.x) >> 5);
  int lane = threadIdx.x & 31;
  if (wid >= E) return;
  int s = src[wid], d = dst[wid];
  float alpha = ex[wid] / (den[d] + 1e-16f);
  const float2* xv = (const float2*)(x2 + (size_t)s * FD);
  float2 v = xv[lane];
  float* ag = aggr + (size_t)d * FD;
  atomicAdd(&ag[lane * 2 + 0], alpha * v.x);
  atomicAdd(&ag[lane * 2 + 1], alpha * v.y);
}

// ---------------- pass D: node_out = x1 + aggr, pooled accumulate -----------
__global__ void gnn_nodeout_kernel(const float* __restrict__ x1,
                                   const float* __restrict__ aggr,
                                   const int* __restrict__ batch,
                                   float* __restrict__ pooled,
                                   float* __restrict__ cnt, int n) {
  int wid  = (int)((blockIdx.x * (long long)blockDim.x + threadIdx.x) >> 5);
  int lane = threadIdx.x & 31;
  if (wid >= n) return;
  int b = batch[wid];
  const float2* av = (const float2*)(x1 + (size_t)wid * FD);
  const float2* gv = (const float2*)(aggr + (size_t)wid * FD);
  float2 o1 = av[lane], o2 = gv[lane];
  atomicAdd(&pooled[b * FD + lane * 2 + 0], o1.x + o2.x);
  atomicAdd(&pooled[b * FD + lane * 2 + 1], o1.y + o2.y);
  if (lane == 0) atomicAdd(&cnt[b], 1.0f);
}

// ---------------- pass E: mean pool ----------------------------------------
__global__ void gnn_finish_kernel(const float* __restrict__ pooled,
                                  const float* __restrict__ cnt,
                                  float* __restrict__ out, int total) {
  int i = blockIdx.x * blockDim.x + threadIdx.x;
  if (i >= total) return;
  out[i] = pooled[i] / fmaxf(cnt[i >> 6], 1.0f);
}

extern "C" void kernel_launch(void* const* d_in, const int* in_sizes, int n_in,
                              void* d_out, int out_size, void* d_ws, size_t ws_size,
                              hipStream_t stream) {
  const float* x    = (const float*)d_in[0];
  const float* ea   = (const float*)d_in[1];
  const float* w1w  = (const float*)d_in[2];  const float* w1b = (const float*)d_in[3];
  const float* w2w  = (const float*)d_in[4];  const float* w2b = (const float*)d_in[5];
  const float* w3w  = (const float*)d_in[6];  const float* w3b = (const float*)d_in[7];
  const float* w4w  = (const float*)d_in[8];  const float* w4b = (const float*)d_in[9];
  const float* w5w  = (const float*)d_in[10]; const float* w5b = (const float*)d_in[11];
  const int*   src  = (const int*)d_in[12];
  const int*   dst  = (const int*)d_in[13];
  const int*   batch= (const int*)d_in[14];
  float* out = (float*)d_out;

  const int N  = in_sizes[0] / FD;
  const int E  = in_sizes[1];
  const int Bg = out_size / FD;

  // ---- workspace layout (floats) ----
  float* ws = (float*)d_ws;
  float* x1 = ws;  ws += (size_t)N * FD;
  float* x2 = ws;  ws += (size_t)N * FD;
  float* x3 = ws;  ws += (size_t)N * FD;
  float* x4 = ws;  ws += (size_t)N * FD;
  float* q5 = ws;  ws += N;
  float* qb = ws;  ws += N;
  float* scores = ws; ws += E;              // reused as exp(score - max)
  float* zbase = ws;                        // everything below zeroed each call
  unsigned* mkey = (unsigned*)ws; ws += N;
  float* den    = ws; ws += N;
  float* aggr   = ws; ws += (size_t)N * FD;
  float* pooled = ws; ws += (size_t)Bg * FD;
  float* cnt    = ws; ws += Bg;
  long long zcount = (long long)(ws - zbase);

  // init accumulators (mkey zero == most-negative ordered key)
  gnn_zero_kernel<<<2048, 256, 0, stream>>>(zbase, zcount);

  // fused WMMA projections
  gnn_proj_kernel<<<(N + 15) / 16, 128, 0, stream>>>(
      x, w1w, w1b, w2w, w2b, w3w, w3b, w4w, w4b, x1, x2, x3, x4, N);

  // per-node scalars for the e5 term
  gnn_q_kernel<<<(N + 255) / 256, 256, 0, stream>>>(x3, w5w, w5b, q5, qb, N);

  // edge passes
  gnn_score_kernel<<<(E + 7) / 8, 256, 0, stream>>>(
      x3, x4, ea, q5, qb, src, dst, scores, mkey, E);
  gnn_exp_kernel<<<(E + 255) / 256, 256, 0, stream>>>(dst, mkey, scores, den, E);
  gnn_scatter_kernel<<<(E + 7) / 8, 256, 0, stream>>>(
      x2, scores, den, src, dst, aggr, E);

  // residual + pooling
  gnn_nodeout_kernel<<<(N + 7) / 8, 256, 0, stream>>>(x1, aggr, batch, pooled, cnt, N);
  gnn_finish_kernel<<<(Bg * FD + 255) / 256, 256, 0, stream>>>(pooled, cnt, out, Bg * FD);
}
